// multimodal_attention_7756710936696
// MI455X (gfx1250) — matrix operand order repaired and checked by reading, not compiled, not run on hardware
//
#include <hip/hip_runtime.h>
#include <hip/hip_bf16.h>
#include <cstdint>

// ---------------------------------------------------------------------------
// Flash attention, B=16, S=2048, D=128, fp32 in/out, f16 WMMA compute.
//
//  * Scores computed transposed (S^T = K * Q^T): f32 C-layout of S^T converts
//    in-lane (zero shuffles) into the f16 A-layout of P for the P*V matmul.
//  * KBLK=64 keys/iteration -> 32 v_wmma_f32_16x16x32_f16 per iteration.
//  * Pack pass writes K (padded f16 rows, stride 136h) and V (per-block
//    transposed f16, stride 72h) into workspace.
//  * Hot loop double-buffers LDS and software-pipelines:
//    wait(prev copies) -> barrier -> issue async copies for block i+1
//    (global_load_async_to_lds_b128, ASYNCcnt) -> WMMA compute on block i.
//  * Fallback instantiation (ws too small) stages from f32 with inline cvt.
// ---------------------------------------------------------------------------

typedef _Float16 v16h __attribute__((ext_vector_type(16)));
typedef _Float16 v8h  __attribute__((ext_vector_type(8)));
typedef float    v8f  __attribute__((ext_vector_type(8)));

#define SEQ      2048
#define DHEAD    128
#define KBLK     64          // keys per iteration (four 16-key WMMA tiles)
#define NBLK     (SEQ / KBLK)
#define QROWS_WG 128         // q rows per workgroup (8 waves * 16 rows)
#define SK_STR   136         // K row stride (halves): 272B, 16B aligned, conflict-free
#define SVT_STR  72          // V^T row stride (halves): 144B, 16B aligned, conflict-free
#define KH_ROW_BYTES   (SK_STR * 2)                  // 272
#define KBLK_BYTES     (KBLK * KH_ROW_BYTES)         // 17408
#define VTBLK_HALVES   (DHEAD * SVT_STR)             // 9216
#define VTBLK_BYTES    (VTBLK_HALVES * 2)            // 18432
#define WS_K_BYTES     ((size_t)16 * SEQ * KH_ROW_BYTES)                  // 8,912,896
#define WS_V_BYTES     ((size_t)16 * NBLK * VTBLK_BYTES)                  // 9,437,184
#define WS_NEED        (WS_K_BYTES + WS_V_BYTES)

// scale * log2(e): softmax done in exp2 domain
#define SCALE_L2E (0.08838834764831845f * 1.4426950408889634f)

// ------------------------- pack pass: f32 -> f16 ---------------------------
__global__ __launch_bounds__(256)
void pack_kv(const float* __restrict__ kg, const float* __restrict__ vg,
             _Float16* __restrict__ kh, _Float16* __restrict__ vt) {
    const int tid  = blockIdx.x * 256 + threadIdx.x;  // 0 .. 262143
    const int rowg = tid >> 3;                        // b*SEQ + s  (0..32767)
    const int c0   = (tid & 7) * 16;                  // d offset

    // K: row-major, padded stride (async-copy source later)
    {
        const float4* k4 = reinterpret_cast<const float4*>(kg + (size_t)rowg * DHEAD + c0);
        float4 a0 = k4[0], a1 = k4[1], a2 = k4[2], a3 = k4[3];
        float ks[16] = {a0.x,a0.y,a0.z,a0.w, a1.x,a1.y,a1.z,a1.w,
                        a2.x,a2.y,a2.z,a2.w, a3.x,a3.y,a3.z,a3.w};
        v8h h0, h1;
#pragma unroll
        for (int i = 0; i < 8; ++i) { h0[i] = (_Float16)ks[i]; h1[i] = (_Float16)ks[i + 8]; }
        _Float16* kd = kh + (size_t)rowg * SK_STR + c0;
        *reinterpret_cast<v8h*>(kd)     = h0;
        *reinterpret_cast<v8h*>(kd + 8) = h1;
    }
    // V: transposed per 64-key block
    {
        const float4* v4 = reinterpret_cast<const float4*>(vg + (size_t)rowg * DHEAD + c0);
        float4 b0 = v4[0], b1 = v4[1], b2 = v4[2], b3 = v4[3];
        float vs[16] = {b0.x,b0.y,b0.z,b0.w, b1.x,b1.y,b1.z,b1.w,
                        b2.x,b2.y,b2.z,b2.w, b3.x,b3.y,b3.z,b3.w};
        const int b    = rowg >> 11;          // / SEQ
        const int srow = rowg & (SEQ - 1);
        const int blk  = srow >> 6;           // / KBLK
        const int key  = srow & (KBLK - 1);
        _Float16* vd = vt + (size_t)(b * NBLK + blk) * VTBLK_HALVES
                          + (size_t)c0 * SVT_STR + key;
#pragma unroll
        for (int j = 0; j < 16; ++j) vd[(size_t)j * SVT_STR] = (_Float16)vs[j];
    }
}

// ------------------------------ attention ----------------------------------
template <bool PACKED>
__global__ __launch_bounds__(256)
void mm_attn_fwd(const float* __restrict__ qg,
                 const float* __restrict__ kg,
                 const float* __restrict__ vg,
                 const _Float16* __restrict__ kh,
                 const _Float16* __restrict__ vt,
                 float* __restrict__ outg) {
    constexpr int NBUF = PACKED ? 2 : 1;                       // double-buffer packed path
    __shared__ __align__(16) _Float16 sK [NBUF][KBLK ][SK_STR ];   // (key, d), f16
    __shared__ __align__(16) _Float16 sVt[NBUF][DHEAD][SVT_STR];   // (d, key), f16

    const int b     = blockIdx.x >> 4;
    const int qbase = (blockIdx.x & 15) * QROWS_WG;
    const int t     = threadIdx.x;
    const int wave  = t >> 5;
    const int lane  = t & 31;
    const int col   = lane & 15;
    const int hi    = lane >> 4;

    const size_t bOff = (size_t)b * SEQ * DHEAD;

    // ---- Q tile once: 4 B-layout fragments (Q^T) --------------------------
    v16h bQ[4];
    {
        const int qrow = qbase + wave * 16 + col;
        const float* qp = qg + bOff + (size_t)qrow * DHEAD;
#pragma unroll
        for (int f = 0; f < 4; ++f) {
            const float4* p4 = reinterpret_cast<const float4*>(qp + f * 32 + hi * 16);
            float4 x0 = p4[0], x1 = p4[1], x2 = p4[2], x3 = p4[3];
            float xs[16] = {x0.x,x0.y,x0.z,x0.w, x1.x,x1.y,x1.z,x1.w,
                            x2.x,x2.y,x2.z,x2.w, x3.x,x3.y,x3.z,x3.w};
            v16h fr;
#pragma unroll
            for (int i = 0; i < 16; ++i) fr[i] = (_Float16)xs[i];
            bQ[f] = fr;
        }
    }

    float m_run = -INFINITY;
    float l_run = 0.0f;
    v8f   oacc[8];
#pragma unroll
    for (int dt = 0; dt < 8; ++dt)
        oacc[dt] = (v8f){0.f,0.f,0.f,0.f,0.f,0.f,0.f,0.f};

    const uint32_t sKlds  = (uint32_t)(uintptr_t)&sK[0][0][0];
    const uint32_t sVtlds = (uint32_t)(uintptr_t)&sVt[0][0][0];
    const int srow = t >> 3;           // 0..31 (staging row, fallback path)
    const int scol = (t & 7) * 16;     // 0..112

    // issue async copies of 64-key block `kb` into LDS buffer `buf`
    auto stage_async = [&](int buf, int kb) {
        const char* ksrc = reinterpret_cast<const char*>(
            kh + ((size_t)b * SEQ + kb) * SK_STR);
        const char* vsrc = reinterpret_cast<const char*>(
            vt + (size_t)(b * NBLK + (kb >> 6)) * VTBLK_HALVES);
        const uint32_t kdst = sKlds  + (uint32_t)buf * KBLK_BYTES;
        const uint32_t vdst = sVtlds + (uint32_t)buf * VTBLK_BYTES;
        for (int i = t; i < KBLK_BYTES / 16; i += 256) {       // 1088 x 16B
            uint64_t ga = (uint64_t)(uintptr_t)ksrc + (uint64_t)i * 16;
            asm volatile("global_load_async_to_lds_b128 %0, %1, off"
                         :: "v"(kdst + i * 16), "v"(ga) : "memory");
        }
        for (int i = t; i < VTBLK_BYTES / 16; i += 256) {      // 1152 x 16B
            uint64_t ga = (uint64_t)(uintptr_t)vsrc + (uint64_t)i * 16;
            asm volatile("global_load_async_to_lds_b128 %0, %1, off"
                         :: "v"(vdst + i * 16), "v"(ga) : "memory");
        }
        if (kb + KBLK < SEQ) {
            __builtin_prefetch(ksrc + KBLK_BYTES + t * 16, 0, 0);
            __builtin_prefetch(vsrc + VTBLK_BYTES + t * 16, 0, 0);
        }
    };

    // WMMA compute for one 64-key block resident in LDS buffer `buf`
    auto compute_block = [&](int buf) {
        // ---- scores: S^T = K * Q^T (four 16-key tiles) --------------------
        v8f st[4];
#pragma unroll
        for (int kt = 0; kt < 4; ++kt) {
            v8f acc = (v8f){0.f,0.f,0.f,0.f,0.f,0.f,0.f,0.f};
#pragma unroll
            for (int f = 0; f < 4; ++f) {
                const int d0 = f * 32 + hi * 16;
                v8h lo = *reinterpret_cast<const v8h*>(&sK[buf][kt * 16 + col][d0]);
                v8h hh = *reinterpret_cast<const v8h*>(&sK[buf][kt * 16 + col][d0 + 8]);
                v16h aK;
#pragma unroll
                for (int i = 0; i < 8; ++i) { aK[i] = lo[i]; aK[i + 8] = hh[i]; }
                acc = __builtin_amdgcn_wmma_f32_16x16x32_f16(
                        false, aK, false, bQ[f], (short)0, acc, false, false);
            }
            st[kt] = acc;
        }

        // ---- online softmax (exp2 domain) ---------------------------------
        float ps[4][8];
        float lmax = -INFINITY;
#pragma unroll
        for (int kt = 0; kt < 4; ++kt)
#pragma unroll
            for (int r = 0; r < 8; ++r) {
                float s = st[kt][r] * SCALE_L2E;
                ps[kt][r] = s;
                lmax = fmaxf(lmax, s);
            }
        lmax = fmaxf(lmax, __shfl_xor(lmax, 16));
        const float m_new = fmaxf(m_run, lmax);
        const float alpha = exp2f(m_run - m_new);

        float psum = 0.0f;
        v16h pA[2];   // A fragments: g covers keys [32g, 32g+32)
#pragma unroll
        for (int g = 0; g < 2; ++g)
#pragma unroll
            for (int r = 0; r < 8; ++r) {
                float p0 = exp2f(ps[2 * g][r]     - m_new);
                float p1 = exp2f(ps[2 * g + 1][r] - m_new);
                psum += p0 + p1;
                pA[g][r]     = (_Float16)p0;
                pA[g][r + 8] = (_Float16)p1;
            }
        psum += __shfl_xor(psum, 16);
        l_run = l_run * alpha + psum;
        m_run = m_new;

        float af[8];
#pragma unroll
        for (int r = 0; r < 8; ++r) af[r] = __shfl(alpha, hi * 8 + r);
#pragma unroll
        for (int dt = 0; dt < 8; ++dt)
#pragma unroll
            for (int r = 0; r < 8; ++r) oacc[dt][r] *= af[r];

        // ---- P * V: 8 d-tiles x 2 key-groups ------------------------------
#pragma unroll
        for (int dt = 0; dt < 8; ++dt) {
#pragma unroll
            for (int g = 0; g < 2; ++g) {
                const _Float16* vtp = &sVt[buf][dt * 16 + col][g * 32 + hi * 8];
                v8h lo = *reinterpret_cast<const v8h*>(vtp);
                v8h hh = *reinterpret_cast<const v8h*>(vtp + 16);
                v16h bV;
#pragma unroll
                for (int i = 0; i < 8; ++i) { bV[i] = lo[i]; bV[i + 8] = hh[i]; }
                oacc[dt] = __builtin_amdgcn_wmma_f32_16x16x32_f16(
                             false, pA[g], false, bV, (short)0, oacc[dt], false, false);
            }
        }
    };

    if constexpr (PACKED) {
        // software-pipelined: copy(i+1) overlaps compute(i)
        stage_async(0, 0);
        for (int it = 0; it < NBLK; ++it) {
            asm volatile("s_wait_asynccnt 0x0" ::: "memory");  // my copies -> buf it&1 done
            __syncthreads();                                   // everyone's done; prev compute done
            if (it + 1 < NBLK) stage_async((it + 1) & 1, (it + 1) * KBLK);
            compute_block(it & 1);
        }
    } else {
        for (int kb = 0; kb < SEQ; kb += KBLK) {
#pragma unroll
            for (int rr = 0; rr < 2; ++rr) {
                const int row = srow + rr * 32;
                const float4* k4 = reinterpret_cast<const float4*>(
                    kg + bOff + (size_t)(kb + row) * DHEAD + scol);
                float4 a0 = k4[0], a1 = k4[1], a2 = k4[2], a3 = k4[3];
                float ks[16] = {a0.x,a0.y,a0.z,a0.w, a1.x,a1.y,a1.z,a1.w,
                                a2.x,a2.y,a2.z,a2.w, a3.x,a3.y,a3.z,a3.w};
                v8h h0, h1;
#pragma unroll
                for (int i = 0; i < 8; ++i) { h0[i] = (_Float16)ks[i]; h1[i] = (_Float16)ks[i + 8]; }
                *reinterpret_cast<v8h*>(&sK[0][row][scol])     = h0;
                *reinterpret_cast<v8h*>(&sK[0][row][scol + 8]) = h1;

                const float4* v4 = reinterpret_cast<const float4*>(
                    vg + bOff + (size_t)(kb + row) * DHEAD + scol);
                float4 b0 = v4[0], b1 = v4[1], b2 = v4[2], b3 = v4[3];
                float vs[16] = {b0.x,b0.y,b0.z,b0.w, b1.x,b1.y,b1.z,b1.w,
                                b2.x,b2.y,b2.z,b2.w, b3.x,b3.y,b3.z,b3.w};
#pragma unroll
                for (int j = 0; j < 16; ++j) sVt[0][scol + j][row] = (_Float16)vs[j];
            }
            __syncthreads();
            compute_block(0);
            __syncthreads();
        }
    }

    // ---- epilogue ---------------------------------------------------------
    const float linv = 1.0f / l_run;
    float lf[8];
#pragma unroll
    for (int r = 0; r < 8; ++r) lf[r] = __shfl(linv, hi * 8 + r);

#pragma unroll
    for (int dt = 0; dt < 8; ++dt) {
        const int d = dt * 16 + col;
#pragma unroll
        for (int r = 0; r < 8; ++r) {
            const int qrow = qbase + wave * 16 + r + 8 * hi;
            outg[bOff + (size_t)qrow * DHEAD + d] = oacc[dt][r] * lf[r];
        }
    }
}

extern "C" void kernel_launch(void* const* d_in, const int* in_sizes, int n_in,
                              void* d_out, int out_size, void* d_ws, size_t ws_size,
                              hipStream_t stream) {
    (void)in_sizes; (void)n_in; (void)out_size;
    const float* q = (const float*)d_in[0];
    const float* k = (const float*)d_in[1];
    const float* v = (const float*)d_in[2];
    float* out = (float*)d_out;

    dim3 grid(16 * (SEQ / QROWS_WG));   // 256 workgroups
    dim3 block(256);                    // 8 waves

    if (ws_size >= WS_NEED && d_ws != nullptr) {
        _Float16* kh = (_Float16*)d_ws;
        _Float16* vt = (_Float16*)((char*)d_ws + WS_K_BYTES);
        pack_kv<<<dim3((16 * SEQ * (DHEAD / 16)) / 256), block, 0, stream>>>(k, v, kh, vt);
        mm_attn_fwd<true><<<grid, block, 0, stream>>>(q, k, v, kh, vt, out);
    } else {
        mm_attn_fwd<false><<<grid, block, 0, stream>>>(q, k, v, nullptr, nullptr, out);
    }
}
